// MSAColAttention_58548994179789
// MI455X (gfx1250) — matrix operand order repaired and checked by reading, not compiled, not run on hardware
//
#include <hip/hip_runtime.h>
#include <stdint.h>

#define NSEQ 256
#define LCOL 256
#define DMSA 256
#define NHEAD 8
#define DHID 32
#define HDIM 256
#define NTOK (NSEQ*LCOL)

typedef __attribute__((ext_vector_type(16))) __bf16 v16bf;
typedef __attribute__((ext_vector_type(8)))  float v8f;
typedef __attribute__((ext_vector_type(4))) unsigned int u32x4;
typedef __attribute__((ext_vector_type(8))) int i32x8;
typedef __attribute__((ext_vector_type(4))) int i32x4;

__device__ __forceinline__ float bf2f(unsigned short s) {
    union { unsigned int u; float f; } t; t.u = ((unsigned int)s) << 16; return t.f;
}
__device__ __forceinline__ unsigned short f2bf(float f) {
    union { float f; unsigned int u; } t; t.f = f;
    unsigned int lsb = (t.u >> 16) & 1u;
    return (unsigned short)((t.u + 0x7FFFu + lsb) >> 16);
}
__device__ __forceinline__ unsigned int packbf(float lo, float hi) {
    return (unsigned int)f2bf(lo) | ((unsigned int)f2bf(hi) << 16);
}

// A-matrix 16x32 bf16 fragment (row-major source): lane m = lane%16, half = lane/16
// slot e -> k = kbase + (e<8 ? half*8+e : 8+half*8+e)  => two contiguous 8-elem runs
__device__ __forceinline__ v16bf frag_a(const unsigned short* row, int kbase, int half) {
    union { uint4 u[2]; v16bf v; } t;
    t.u[0] = *(const uint4*)(row + kbase + half * 8);
    t.u[1] = *(const uint4*)(row + kbase + 16 + half * 8);
    return t.v;
}
// B-matrix 32x16 bf16 fragment: lane n = lane%16 selects column (pre-transposed source row),
// slot e -> k = kbase + (e<8 ? half*8+e : 8+half*8+e)  => two contiguous 8-elem runs
__device__ __forceinline__ v16bf frag_b(const unsigned short* col, int kbase, int half) {
    union { uint4 u[2]; v16bf v; } t;
    t.u[0] = *(const uint4*)(col + kbase + half * 8);
    t.u[1] = *(const uint4*)(col + kbase + 16 + half * 8);
    return t.v;
}

// ---------------- Kernel 1: weight transpose + f32->bf16 cast ----------------
__global__ void k_prep_w(const float* Wq, const float* Wk, const float* Wv,
                         const float* Wg, const float* Wo, unsigned short* Wt) {
    int m = blockIdx.y;
    const float* W = (m == 0) ? Wq : (m == 1) ? Wk : (m == 2) ? Wv : (m == 3) ? Wg : Wo;
    unsigned short* out = Wt + (size_t)m * 256 * 256;
    int idx = blockIdx.x * blockDim.x + threadIdx.x;
    int n = idx >> 8, k = idx & 255;
    out[n * 256 + k] = f2bf(W[k * 256 + n]);  // Wt[n][k] = W[k][n]
}

// ---------------- Kernel 2: LayerNorm -> bf16 ----------------
__global__ void k_ln(const float* msa, const float* g, const float* b, unsigned short* Xb) {
    int wave = threadIdx.x >> 5;
    int lane = threadIdx.x & 31;
    size_t tok = (size_t)blockIdx.x * 8 + wave;
    const float* row = msa + tok * DMSA;
    float v[8], s = 0.f, s2 = 0.f;
#pragma unroll
    for (int j = 0; j < 8; j++) { v[j] = row[lane + j * 32]; s += v[j]; s2 += v[j] * v[j]; }
#pragma unroll
    for (int off = 16; off >= 1; off >>= 1) {
        s += __shfl_xor(s, off, 32); s2 += __shfl_xor(s2, off, 32);
    }
    float mu = s * (1.0f / 256.0f);
    float var = s2 * (1.0f / 256.0f) - mu * mu;
    float rs = rsqrtf(var + 1e-5f);
    unsigned short* orow = Xb + tok * DMSA;
#pragma unroll
    for (int j = 0; j < 8; j++) {
        int c = lane + j * 32;
        orow[c] = f2bf((v[j] - mu) * rs * g[c] + b[c]);
    }
}

// ---------------- Kernel 3: projections (Q,K,V,sigmoid-gate) via WMMA ----------------
__global__ void k_proj(const unsigned short* Xb, const unsigned short* Wt, const float* bg,
                       unsigned short* Q, unsigned short* K, unsigned short* V, unsigned short* G) {
    int which = blockIdx.z;
    const unsigned short* W = Wt + (size_t)which * 65536;
    unsigned short* out = (which == 0) ? Q : (which == 1) ? K : (which == 2) ? V : G;
    int wave = threadIdx.x >> 5, lane = threadIdx.x & 31;
    int tile = blockIdx.x * 8 + wave;         // 4096 M-tiles x 16 N-tiles
    int mbase = (tile >> 4) * 16, nbase = (tile & 15) * 16;
    int half = lane >> 4, ln = lane & 15;
    const unsigned short* arow = Xb + (size_t)(mbase + ln) * DMSA;
    const unsigned short* bcol = W + (size_t)(nbase + ln) * DMSA;
    v8f c = {};
#pragma unroll
    for (int kk = 0; kk < 8; kk++) {
        v16bf a = frag_a(arow, kk * 32, half);
        v16bf bb = frag_b(bcol, kk * 32, half);
        c = __builtin_amdgcn_wmma_f32_16x16x32_bf16(false, a, false, bb, (short)0, c, false, false);
    }
    if (which < 3) {
#pragma unroll
        for (int r = 0; r < 8; r++) {
            int m = mbase + r + 8 * half, n = nbase + ln;
            out[(size_t)m * HDIM + n] = f2bf(c[r]);
        }
    } else {
#pragma unroll
        for (int r = 0; r < 8; r++) {
            int m = mbase + r + 8 * half, n = nbase + ln;
            float x = c[r] + bg[n];
            out[(size_t)m * HDIM + n] = f2bf(1.0f / (1.0f + __expf(-x)));
        }
    }
}

// ---------------- Kernel 4: per-(column,head) attention over sequence axis ----------------
// Computes S^T = K @ Q^T so the C-layout lane index is the query and each accumulator
// register holds consecutive keys -> row-major P with packed b128 LDS stores.
__global__ __launch_bounds__(128) void k_attn(const unsigned short* Q, const unsigned short* K,
                                              const unsigned short* V, unsigned short* O) {
    __shared__ unsigned short Klds[256][32];    // K_ih row-major       (16 KB) -- TDM target
    __shared__ unsigned short Vtlds[32][256];   // V_ih transposed      (16 KB)
    __shared__ unsigned short Plds[4][16][256]; // per-wave P strip     (32 KB)
    int i = blockIdx.x, h = blockIdx.y;
    int wave = threadIdx.x >> 5, lane = threadIdx.x & 31;
    int half = lane >> 4, ln = lane & 15;
    const float scale = 0.17677669529663687f;   // 1/sqrt(32)

#if __has_builtin(__builtin_amdgcn_tensor_load_to_lds)
    // ---- Tensor Data Mover: DMA the 256x32 bf16 K tile (rows strided 128 KB) into LDS ----
    if (wave == 0) {
        unsigned long long ga =
            (unsigned long long)(const void*)(K + ((size_t)i) * HDIM + (size_t)h * DHID);
        unsigned int ldsoff = (unsigned int)(unsigned long long)(void*)&Klds[0][0];
        u32x4 g0 = { 1u,                                    // count=1 valid user descriptor
                     ldsoff,                                // lds_addr [63:32]
                     (unsigned int)(ga & 0xffffffffu),      // global_addr [95:64]
                     (unsigned int)((ga >> 32) & 0x01ffffffu) | (2u << 30) }; // addr hi + type=2
        i32x8 g1 = { (int)0x00010000,      // data_size=2B, no multicast/pad/iterate
                     (int)(32u  << 16),    // tensor_dim0 = 32 elements
                     (int)(256u << 16),    // tensor_dim1 = 256 rows
                     (int)(32u  << 16),    // tile_dim0 = 32
                     (int)0x00000100,      // tile_dim1 = 256, tile_dim2 = 0
                     (int)(LCOL * HDIM),   // tensor_dim0_stride = 65536 elements
                     0, 0 };
        i32x4 z4 = { 0, 0, 0, 0 };
#if defined(__clang_major__) && (__clang_major__ >= 23)
        i32x8 z8 = { 0, 0, 0, 0, 0, 0, 0, 0 };
        __builtin_amdgcn_tensor_load_to_lds(g0, g1, z4, z4, z8, 0);
#else
        __builtin_amdgcn_tensor_load_to_lds(g0, g1, z4, z4, 0);
#endif
        __builtin_amdgcn_s_wait_tensorcnt(0);
    }
#else
    // fallback: manual K staging
#pragma unroll
    for (int rr = 0; rr < 2; rr++) {
        int row = threadIdx.x + rr * 128;
        const uint4* kp = (const uint4*)(K + ((size_t)row * LCOL + i) * HDIM + h * DHID);
#pragma unroll
        for (int q4 = 0; q4 < 4; q4++) *(uint4*)&Klds[row][q4 * 8] = kp[q4];
    }
#endif

    // stage V^T manually (TDM cannot transpose): thread t handles rows t, t+128
#pragma unroll
    for (int rr = 0; rr < 2; rr++) {
        int row = threadIdx.x + rr * 128;
        union { uint4 u[4]; unsigned short s[32]; } vv;
        const uint4* vp = (const uint4*)(V + ((size_t)row * LCOL + i) * HDIM + h * DHID);
#pragma unroll
        for (int q4 = 0; q4 < 4; q4++) vv.u[q4] = vp[q4];
#pragma unroll
        for (int d = 0; d < 32; d++) Vtlds[d][row] = vv.s[d];
    }
    __syncthreads();

    for (int j = 0; j < 4; j++) {           // each wave: 4 query tiles
        int qbase = (wave * 4 + j) * 16;
        // B operand = Q^T tile: lane = query column, one contiguous 32-elem run of d
        const unsigned short* qrow = Q + ((size_t)(qbase + ln) * LCOL + i) * HDIM + h * DHID;
        v16bf bq = frag_b(qrow, 0, half);
        v8f S[16];                           // S^T tiles: m = key_local, n = query
#pragma unroll
        for (int kt = 0; kt < 16; kt++) {
            v16bf ak = frag_a(&Klds[kt * 16 + ln][0], 0, half);
            v8f z = {};
            S[kt] = __builtin_amdgcn_wmma_f32_16x16x32_bf16(false, ak, false, bq, (short)0, z, false, false);
        }
        // softmax over 256 keys: each lane owns one query, keys live in (r, kt, half)
        float mx = -1e30f;
#pragma unroll
        for (int kt = 0; kt < 16; kt++)
#pragma unroll
            for (int r = 0; r < 8; r++) mx = fmaxf(mx, S[kt][r]);
        mx = fmaxf(mx, __shfl_xor(mx, 16, 32));   // combine halves (wave32)
        float sm = 0.f;
#pragma unroll
        for (int kt = 0; kt < 16; kt++)
#pragma unroll
            for (int r = 0; r < 8; r++) {
                float p = __expf((S[kt][r] - mx) * scale);
                S[kt][r] = p; sm += p;
            }
        sm += __shfl_xor(sm, 16, 32);
        float inv = 1.0f / sm;
        // P -> LDS row-major: 8 consecutive keys per register -> one b128 store per tile
#pragma unroll
        for (int kt = 0; kt < 16; kt++) {
            uint4 pk;
            pk.x = packbf(S[kt][0] * inv, S[kt][1] * inv);
            pk.y = packbf(S[kt][2] * inv, S[kt][3] * inv);
            pk.z = packbf(S[kt][4] * inv, S[kt][5] * inv);
            pk.w = packbf(S[kt][6] * inv, S[kt][7] * inv);
            *(uint4*)&Plds[wave][ln][kt * 16 + 8 * half] = pk;
        }
        // O = P @ V : contraction over 256 keys in 8 chunks of 32
        v8f acc0 = {}, acc1 = {};
#pragma unroll
        for (int kc = 0; kc < 8; kc++) {
            v16bf ap = frag_a(&Plds[wave][ln][0], kc * 32, half);
            v16bf b0 = frag_b(&Vtlds[ln][0], kc * 32, half);
            v16bf b1 = frag_b(&Vtlds[16 + ln][0], kc * 32, half);
            acc0 = __builtin_amdgcn_wmma_f32_16x16x32_bf16(false, ap, false, b0, (short)0, acc0, false, false);
            acc1 = __builtin_amdgcn_wmma_f32_16x16x32_bf16(false, ap, false, b1, (short)0, acc1, false, false);
        }
#pragma unroll
        for (int r = 0; r < 8; r++) {
            int qn = qbase + r + 8 * half;
            size_t obase = ((size_t)qn * LCOL + i) * HDIM + h * DHID;
            O[obase + ln] = f2bf(acc0[r]);
            O[obase + 16 + ln] = f2bf(acc1[r]);
        }
    }
}

// ---------------- Kernel 5: Y = (G .* O) @ Wo + bo (f32 out) ----------------
__global__ void k_out(const unsigned short* Oin, const unsigned short* G,
                      const unsigned short* Wot, const float* bo, float* Y) {
    int wave = threadIdx.x >> 5, lane = threadIdx.x & 31;
    int tile = blockIdx.x * 8 + wave;
    int mbase = (tile >> 4) * 16, nbase = (tile & 15) * 16;
    int half = lane >> 4, ln = lane & 15;
    const unsigned short* orow = Oin + (size_t)(mbase + ln) * HDIM;
    const unsigned short* grow = G + (size_t)(mbase + ln) * HDIM;
    const unsigned short* bcol = Wot + (size_t)(nbase + ln) * HDIM;
    v8f c = {};
#pragma unroll
    for (int kk = 0; kk < 8; kk++) {
        union { uint4 u[2]; unsigned short s[16]; } o_, g_;
        o_.u[0] = *(const uint4*)(orow + kk * 32 + half * 8);
        o_.u[1] = *(const uint4*)(orow + kk * 32 + 16 + half * 8);
        g_.u[0] = *(const uint4*)(grow + kk * 32 + half * 8);
        g_.u[1] = *(const uint4*)(grow + kk * 32 + 16 + half * 8);
        union { unsigned int u[8]; v16bf v; } a;
#pragma unroll
        for (int e = 0; e < 8; e++)
            a.u[e] = packbf(bf2f(o_.s[2 * e]) * bf2f(g_.s[2 * e]),
                            bf2f(o_.s[2 * e + 1]) * bf2f(g_.s[2 * e + 1]));
        v16bf bb = frag_b(bcol, kk * 32, half);
        c = __builtin_amdgcn_wmma_f32_16x16x32_bf16(false, a.v, false, bb, (short)0, c, false, false);
    }
#pragma unroll
    for (int r = 0; r < 8; r++) {
        int m = mbase + r + 8 * half, n = nbase + ln;
        Y[(size_t)m * DMSA + n] = c[r] + bo[n];
    }
}

extern "C" void kernel_launch(void* const* d_in, const int* in_sizes, int n_in,
                              void* d_out, int out_size, void* d_ws, size_t ws_size,
                              hipStream_t stream) {
    (void)in_sizes; (void)n_in; (void)out_size; (void)ws_size;
    const float* msa  = (const float*)d_in[0];
    const float* ln_g = (const float*)d_in[1];
    const float* ln_b = (const float*)d_in[2];
    const float* Wq   = (const float*)d_in[3];
    const float* Wk   = (const float*)d_in[4];
    const float* Wv   = (const float*)d_in[5];
    const float* Wg   = (const float*)d_in[6];
    const float* bg   = (const float*)d_in[7];
    const float* Wo   = (const float*)d_in[8];
    const float* bo   = (const float*)d_in[9];
    float* Y = (float*)d_out;

    char* ws = (char*)d_ws;
    size_t sz = (size_t)NTOK * 256 * sizeof(unsigned short); // 32 MB per tensor
    unsigned short* Xb = (unsigned short*)(ws);
    unsigned short* Qb = (unsigned short*)(ws + 1 * sz);
    unsigned short* Kb = (unsigned short*)(ws + 2 * sz);
    unsigned short* Vb = (unsigned short*)(ws + 3 * sz);
    unsigned short* Gb = (unsigned short*)(ws + 4 * sz);
    unsigned short* Ob = (unsigned short*)(ws + 5 * sz);
    unsigned short* Wt = (unsigned short*)(ws + 6 * sz);  // 5 x 128 KB transposed bf16 weights

    hipLaunchKernelGGL(k_prep_w, dim3(256, 5), dim3(256), 0, stream, Wq, Wk, Wv, Wg, Wo, Wt);
    hipLaunchKernelGGL(k_ln,     dim3(8192),   dim3(256), 0, stream, msa, ln_g, ln_b, Xb);
    hipLaunchKernelGGL(k_proj,   dim3(8192, 1, 4), dim3(256), 0, stream, Xb, Wt, bg, Qb, Kb, Vb, Gb);
    hipLaunchKernelGGL(k_attn,   dim3(256, 8), dim3(128), 0, stream, Qb, Kb, Vb, Ob);
    hipLaunchKernelGGL(k_out,    dim3(8192),   dim3(256), 0, stream, Ob, Gb, Wt + 4 * 65536, bo, Y);
}
